// LinearNO_2765958939221
// MI455X (gfx1250) — matrix operand order repaired and checked by reading, not compiled, not run on hardware
//
#include <hip/hip_runtime.h>

typedef __attribute__((ext_vector_type(16))) __bf16       v16bf;
typedef __attribute__((ext_vector_type(8)))  float        v8f;
typedef __attribute__((ext_vector_type(8)))  unsigned int v8u;
typedef __attribute__((ext_vector_type(4)))  float        f4;
typedef __attribute__((ext_vector_type(4)))  unsigned int u4;
typedef __attribute__((ext_vector_type(2)))  unsigned int u2;

#define SEQ     8192
#define NB      8          // batch
#define CH      512
#define NHEADS  8
#define DH      64
#define SL      32
#define ROWS    (NB * SEQ)   // 65536
#define QKVCOLS 1024         // [q:256 | k:256 | v:512]

constexpr int AS = 40;       // LDS row stride (ushorts): 80 B, 16 B aligned
constexpr int BS = 40;

// native bf16 convert (lowers to v_cvt_*bf16_f32 on gfx1250)
__device__ __forceinline__ unsigned short f2bf(float f) {
  __bf16 h = (__bf16)f;
  return __builtin_bit_cast(unsigned short, h);
}
__device__ __forceinline__ float bf2f(unsigned short h) {
  unsigned int u = ((unsigned int)h) << 16;
  return __builtin_bit_cast(float, u);
}
__device__ __forceinline__ unsigned int pk2(float lo, float hi) {
  return (unsigned int)f2bf(lo) | ((unsigned int)f2bf(hi) << 16);
}

// ---------------------------------------------------------------- zero
__global__ void zero_f32(float* p, int n) {
  int i = blockIdx.x * blockDim.x + threadIdx.x;
  if (i < n) p[i] = 0.f;
}

// ------------------------------------------- fold W_in into per-head q/k/v
// Wc: [512][1024] bf16, cols 0..255 = q (h*32+s), 256..511 = k, 512..1023 = v (h*64+c)
__global__ __launch_bounds__(256)
void build_weights(const float* __restrict__ W_in, const float* __restrict__ b_in,
                   const float* __restrict__ Wq, const float* __restrict__ Wk,
                   const float* __restrict__ Wv,
                   unsigned short* __restrict__ Wc, float* __restrict__ bias_c) {
  int bid = blockIdx.x;
  if (bid < 2048) {
    int id  = bid * 256 + threadIdx.x;   // 0..524287
    int col = id & 1023;
    int r   = id >> 10;                  // 0..511
    const float* xr = W_in + (long)r * CH;
    float acc = 0.f;
    if (col < 256) {
      int h = col >> 5, s = col & 31;
      for (int d = 0; d < DH; ++d) acc += xr[h * DH + d] * Wq[d * SL + s];
    } else if (col < 512) {
      int c2 = col - 256; int h = c2 >> 5, s = c2 & 31;
      for (int d = 0; d < DH; ++d) acc += xr[h * DH + d] * Wk[d * SL + s];
    } else {
      int c2 = col - 512; int h = c2 >> 6, c = c2 & 63;
      for (int d = 0; d < DH; ++d) acc += xr[h * DH + d] * Wv[d * DH + c];
    }
    Wc[id] = f2bf(acc);
  } else {                                // folded biases (4 blocks)
    int col = (bid - 2048) * 256 + threadIdx.x;
    if (col >= 1024) return;
    float acc = 0.f;
    if (col < 256) {
      int h = col >> 5, s = col & 31;
      for (int d = 0; d < DH; ++d) acc += b_in[h * DH + d] * Wq[d * SL + s];
    } else if (col < 512) {
      int c2 = col - 256; int h = c2 >> 5, s = c2 & 31;
      for (int d = 0; d < DH; ++d) acc += b_in[h * DH + d] * Wk[d * SL + s];
    } else {
      int c2 = col - 512; int h = c2 >> 6, c = c2 & 63;
      for (int d = 0; d < DH; ++d) acc += b_in[h * DH + d] * Wv[d * DH + c];
    }
    bias_c[col] = acc;
  }
}

// ------------------------------ GEMM staging helpers (vectorized b128 loads)
template <bool A_BF16>
__device__ __forceinline__ void stage_gload(const float* __restrict__ Af,
                                            const unsigned short* __restrict__ Ab,
                                            const unsigned short* __restrict__ Bg,
                                            long m0, int n0, int K, int N, int kk, int t,
                                            f4 (&ar)[4], u4 (&abr)[2], u4 (&br)[2]) {
  if (!A_BF16) {
#pragma unroll
    for (int i = 0; i < 4; ++i) {
      int vi = t + i * 256;                   // 0..1023 float4s of a 128x32 tile
      int r = vi >> 3, c4 = vi & 7;
      ar[i] = *(const f4*)(Af + (m0 + r) * (long)K + kk + c4 * 4);
    }
  } else {
#pragma unroll
    for (int i = 0; i < 2; ++i) {
      int vi = t + i * 256;                   // 0..511 8-elem vecs
      int r = vi >> 2, c8 = vi & 3;
      abr[i] = *(const u4*)(Ab + (m0 + r) * (long)K + kk + c8 * 8);
    }
  }
#pragma unroll
  for (int i = 0; i < 2; ++i) {
    int vi = t + i * 256;                     // 0..511 8-elem vecs of 32x128 tile
    int kr = vi >> 4, nc = (vi & 15) * 8;
    br[i] = *(const u4*)(Bg + (long)(kk + kr) * N + n0 + nc);
  }
}

template <bool A_BF16>
__device__ __forceinline__ void stage_store(unsigned short* __restrict__ a_sm,
                                            unsigned short* __restrict__ b_sm, int t,
                                            const f4 (&ar)[4], const u4 (&abr)[2],
                                            const u4 (&br)[2]) {
  if (!A_BF16) {
#pragma unroll
    for (int i = 0; i < 4; ++i) {
      int vi = t + i * 256;
      int r = vi >> 3, c4 = vi & 7;
      u2 p;
      p[0] = pk2(ar[i][0], ar[i][1]);
      p[1] = pk2(ar[i][2], ar[i][3]);
      *(u2*)(a_sm + r * AS + c4 * 4) = p;     // 8 B aligned
    }
  } else {
#pragma unroll
    for (int i = 0; i < 2; ++i) {
      int vi = t + i * 256;
      int r = vi >> 2, c8 = vi & 3;
      *(u4*)(a_sm + r * AS + c8 * 8) = abr[i];  // 16 B aligned
    }
  }
#pragma unroll
  for (int i = 0; i < 2; ++i) {
    int vi = t + i * 256;
    int kr = vi >> 4, nc = (vi & 15) * 8;
    const unsigned short* s = (const unsigned short*)&br[i];
#pragma unroll
    for (int j = 0; j < 8; ++j)               // transpose scatter -> [n][k]
      b_sm[(nc + j) * BS + kr] = s[j];
  }
}

// ---------------------------------------------------------------- WMMA GEMM
// D[m,n] = sum_k A[m,k]*B[k,n] + bias[n].  128x128 tile, 8 waves (4Mx2N),
// K-step 32, double-buffered LDS, software-pipelined b128 global loads.
template <bool A_BF16, bool OUT_BF16>
__global__ __launch_bounds__(256)
void gemm_wmma_bf16(const void* __restrict__ Ap, const unsigned short* __restrict__ Bp,
                    const float* __restrict__ bias, void* __restrict__ Dp,
                    int N, int K, long strideA, long strideB, long strideD) {
  __shared__ __align__(16) unsigned short a_sm[2][128 * AS];
  __shared__ __align__(16) unsigned short b_sm[2][128 * BS];

  const int  zb = blockIdx.z;
  const int  n0 = blockIdx.x * 128;
  const long m0 = (long)blockIdx.y * 128;
  const int  t    = threadIdx.x;
  const int  lane = t & 31, w = t >> 5;
  const int  wm = w & 3, wn = w >> 2;    // 4 M-groups x 2 N-groups of waves
  const int  l15 = lane & 15, hi = lane >> 4;

  const float*          Af = (const float*)Ap + strideA * zb;
  const unsigned short* Ab = (const unsigned short*)Ap + strideA * zb;
  const unsigned short* Bg = Bp + strideB * zb;

  v8f acc[2][4];
#pragma unroll
  for (int i = 0; i < 2; ++i)
#pragma unroll
    for (int j = 0; j < 4; ++j)
      acc[i][j] = (v8f){0.f, 0.f, 0.f, 0.f, 0.f, 0.f, 0.f, 0.f};

  f4 ar[4]; u4 abr[2]; u4 br[2];
  stage_gload<A_BF16>(Af, Ab, Bg, m0, n0, K, N, 0, t, ar, abr, br);
  stage_store<A_BF16>(a_sm[0], b_sm[0], t, ar, abr, br);
  __syncthreads();

  const int nk = K >> 5;
  for (int kt = 0; kt < nk; ++kt) {
    const int cur = kt & 1;
    if (kt + 1 < nk)        // issue next tile's global loads early
      stage_gload<A_BF16>(Af, Ab, Bg, m0, n0, K, N, (kt + 1) * 32, t, ar, abr, br);
    if (kt + 2 < nk) {      // prefetch tile after next
      int kk2 = (kt + 2) * 32;
      if (!A_BF16) __builtin_prefetch(Af + (m0 + (t >> 3)) * (long)K + kk2 + (t & 7) * 4, 0, 0);
      else         __builtin_prefetch(Ab + (m0 + (t >> 2)) * (long)K + kk2 + (t & 3) * 8, 0, 0);
      __builtin_prefetch(Bg + (long)(kk2 + (t >> 4)) * N + n0 + (t & 15) * 8, 0, 0);
    }

    v16bf afrag[2], bfrag[4];
#pragma unroll
    for (int mt = 0; mt < 2; ++mt) {
      const unsigned short* ab = &a_sm[cur][(wm * 32 + mt * 16 + l15) * AS];
      v8u u;
#pragma unroll
      for (int p = 0; p < 4; ++p) u[p] = *(const unsigned int*)(ab + hi * 8 + 2 * p);
#pragma unroll
      for (int p = 4; p < 8; ++p) u[p] = *(const unsigned int*)(ab + 8 + hi * 8 + 2 * p);
      afrag[mt] = __builtin_bit_cast(v16bf, u);
    }
#pragma unroll
    for (int nt = 0; nt < 4; ++nt) {
      const unsigned short* bb = &b_sm[cur][(wn * 64 + nt * 16 + l15) * BS];
      v8u u;
#pragma unroll
      for (int p = 0; p < 8; ++p) u[p] = *(const unsigned int*)(bb + (p < 4 ? hi * 8 : 8 + hi * 8) + 2 * p);
      bfrag[nt] = __builtin_bit_cast(v16bf, u);
    }
#pragma unroll
    for (int mt = 0; mt < 2; ++mt)
#pragma unroll
      for (int nt = 0; nt < 4; ++nt)
        acc[mt][nt] = __builtin_amdgcn_wmma_f32_16x16x32_bf16(
            false, afrag[mt], false, bfrag[nt], (short)0, acc[mt][nt], false, false);

    if (kt + 1 < nk)        // write next tile into the other buffer
      stage_store<A_BF16>(a_sm[cur ^ 1], b_sm[cur ^ 1], t, ar, abr, br);
    __syncthreads();
  }

  // epilogue: pointer-strided stores, bias hoisted
#pragma unroll
  for (int mt = 0; mt < 2; ++mt) {
    const long row0 = m0 + wm * 32 + mt * 16 + hi * 8;
#pragma unroll
    for (int nt = 0; nt < 4; ++nt) {
      const int n = n0 + wn * 64 + nt * 16 + l15;
      const float bv = bias[n];
      if (OUT_BF16) {
        unsigned short* p = (unsigned short*)Dp + strideD * zb + row0 * (long)N + n;
#pragma unroll
        for (int r = 0; r < 8; ++r) { *p = f2bf(acc[mt][nt][r] + bv); p += N; }
      } else {
        float* p = (float*)Dp + strideD * zb + row0 * (long)N + n;
#pragma unroll
        for (int r = 0; r < 8; ++r) { *p = acc[mt][nt][r] + bv; p += N; }
      }
    }
  }
}

// ------------------------------------- k softmax column stats (max, sum-exp)
__global__ __launch_bounds__(256)
void k_stats(const unsigned short* __restrict__ QKV, float* __restrict__ kmax,
             float* __restrict__ ksum) {
  int cg = blockIdx.x, b = blockIdx.y;
  int t = threadIdx.x, c = t & 31, rg = t >> 5;
  int col = cg * 32 + c;                                   // 0..255
  const unsigned short* base = QKV + (long)b * SEQ * QKVCOLS + 256 + col;
  __shared__ float red[8][32];
  float m = -3.4e38f;
  for (int n = rg; n < SEQ; n += 8) m = fmaxf(m, bf2f(base[(long)n * QKVCOLS]));
  red[rg][c] = m; __syncthreads();
  if (rg == 0) { for (int i = 1; i < 8; ++i) m = fmaxf(m, red[i][c]); red[0][c] = m; }
  __syncthreads();
  m = red[0][c];
  __syncthreads();
  float s = 0.f;
  for (int n = rg; n < SEQ; n += 8) s += __expf(bf2f(base[(long)n * QKVCOLS]) - m);
  red[rg][c] = s; __syncthreads();
  if (rg == 0) {
    for (int i = 1; i < 8; ++i) s += red[i][c];
    kmax[b * 256 + col] = m;
    ksum[b * 256 + col] = s;
  }
}

// ------------------------- kv[b,h,s,c] += sum_n exp(k - kmax) * v (split-K)
__global__ __launch_bounds__(256)
void kv_accum(const unsigned short* __restrict__ QKV, const float* __restrict__ kmax,
              float* __restrict__ kv_raw) {
  int split = blockIdx.x;            // 0..31
  int bh = blockIdx.y;               // 0..63
  int b = bh >> 3, h = bh & 7;
  int t = threadIdx.x;
  int s = t >> 3, cb = (t & 7) * 8;
  __shared__ float kls[8][32];
  __shared__ float vls[8][64];
  float acc[8] = {0.f, 0.f, 0.f, 0.f, 0.f, 0.f, 0.f, 0.f};
  const unsigned short* rowbase = QKV + (long)b * SEQ * QKVCOLS;
  const float* km = kmax + b * 256 + h * 32;
  int n0 = split * (SEQ / 32);
  for (int chunk = 0; chunk < 32; ++chunk) {
    int nb = n0 + chunk * 8;
#pragma unroll
    for (int i = 0; i < 3; ++i) {
      int e = t + i * 256;           // 0..767
      int r = e / 96, c = e % 96;
      const unsigned short* rp = rowbase + (long)(nb + r) * QKVCOLS;
      if (c < 32) kls[r][c]      = __expf(bf2f(rp[256 + h * 32 + c]) - km[c]);
      else        vls[r][c - 32] = bf2f(rp[512 + h * 64 + (c - 32)]);
    }
    __syncthreads();
#pragma unroll
    for (int r = 0; r < 8; ++r) {
      float ek = kls[r][s];
#pragma unroll
      for (int i = 0; i < 8; ++i) acc[i] += ek * vls[r][cb + i];
    }
    __syncthreads();
  }
  float* kvp = kv_raw + ((long)bh * 32 + s) * 64 + cb;
#pragma unroll
  for (int i = 0; i < 8; ++i) atomicAdd(&kvp[i], acc[i]);
}

// ------------------------------------------- per-row q softmax (32 per head)
__global__ __launch_bounds__(256)
void soft_q(const unsigned short* __restrict__ QKV, unsigned short* __restrict__ softQ) {
  long id = (long)blockIdx.x * 256 + threadIdx.x;  // 0..524287
  long row = id >> 3; int h = (int)(id & 7);
  const unsigned short* qp = QKV + row * QKVCOLS + h * 32;
  float v[32]; float m = -3.4e38f;
#pragma unroll
  for (int s = 0; s < 32; ++s) { v[s] = bf2f(qp[s]); m = fmaxf(m, v[s]); }
  float sum = 0.f;
#pragma unroll
  for (int s = 0; s < 32; ++s) { v[s] = __expf(v[s] - m); sum += v[s]; }
  float inv = 1.f / sum;
  unsigned short* op = softQ + row * 256 + h * 32;
#pragma unroll
  for (int s = 0; s < 32; ++s) op[s] = f2bf(v[s] * inv);
}

// -------------------- Wb[b] = blockdiag(kv_norm_h) @ W_out   [8][256][512]
__global__ __launch_bounds__(256)
void build_wb(const float* __restrict__ kv_raw, const float* __restrict__ ksum,
              const float* __restrict__ W_out, unsigned short* __restrict__ Wb) {
  long id = (long)blockIdx.x * 256 + threadIdx.x;  // 0..1048575
  int o  = (int)(id % 512);
  int rs = (int)((id / 512) % 256);
  int b  = (int)(id / (512 * 256));
  int h = rs >> 5, s = rs & 31;
  const float* kvp = kv_raw + (((long)(b * 8 + h)) * 32 + s) * 64;
  const float* wp  = W_out + (long)(h * 64) * 512 + o;
  float acc = 0.f;
#pragma unroll
  for (int c = 0; c < 64; ++c) acc += kvp[c] * wp[(long)c * 512];
  Wb[id] = f2bf(acc / ksum[b * 256 + rs]);
}

extern "C" void kernel_launch(void* const* d_in, const int* in_sizes, int n_in,
                              void* d_out, int out_size, void* d_ws, size_t ws_size,
                              hipStream_t stream) {
  (void)in_sizes; (void)n_in; (void)out_size; (void)ws_size;
  const float* x     = (const float*)d_in[0];
  const float* W_in  = (const float*)d_in[1];
  const float* b_in  = (const float*)d_in[2];
  const float* Wq    = (const float*)d_in[3];
  const float* Wk    = (const float*)d_in[4];
  const float* Wv    = (const float*)d_in[5];
  const float* W_out = (const float*)d_in[6];
  const float* b_out = (const float*)d_in[7];

  char* ws = (char*)d_ws;
  size_t off = 0;
  auto alloc = [&](size_t bytes) {
    void* p = ws + off;
    off = (off + bytes + 255) & ~(size_t)255;
    return p;
  };
  unsigned short* Wc    = (unsigned short*)alloc((size_t)512 * 1024 * 2);
  float*          bias_c= (float*)alloc(1024 * 4);
  unsigned short* QKV   = (unsigned short*)alloc((size_t)ROWS * QKVCOLS * 2);
  float*          kmax  = (float*)alloc(2048 * 4);
  float*          ksum  = (float*)alloc(2048 * 4);
  float*          kv_raw= (float*)alloc((size_t)131072 * 4);
  unsigned short* softQ = (unsigned short*)alloc((size_t)ROWS * 256 * 2);
  unsigned short* Wb    = (unsigned short*)alloc((size_t)NB * 256 * 512 * 2);

  zero_f32<<<512, 256, 0, stream>>>(kv_raw, 131072);
  build_weights<<<2052, 256, 0, stream>>>(W_in, b_in, Wq, Wk, Wv, Wc, bias_c);
  // QKV = x @ Wc + bias_c  (M=65536, N=1024, K=512) -> bf16
  gemm_wmma_bf16<false, true><<<dim3(8, 512, 1), 256, 0, stream>>>(
      x, Wc, bias_c, QKV, 1024, 512, 0L, 0L, 0L);
  k_stats<<<dim3(8, NB), 256, 0, stream>>>(QKV, kmax, ksum);
  kv_accum<<<dim3(32, 64), 256, 0, stream>>>(QKV, kmax, kv_raw);
  soft_q<<<2048, 256, 0, stream>>>(QKV, softQ);
  build_wb<<<4096, 256, 0, stream>>>(kv_raw, ksum, W_out, Wb);
  // out[b] = softQ[b] @ Wb[b] + b_out  (M=8192, N=512, K=256, batched b=8) -> fp32
  gemm_wmma_bf16<true, false><<<dim3(4, 64, NB), 256, 0, stream>>>(
      softQ, Wb, b_out, d_out, 512, 256, (long)SEQ * 256, (long)256 * 512, (long)SEQ * 512);
}
